// UltraNet_FixQ_79379585564681
// MI455X (gfx1250) — 8-bit matrix operand order repaired and checked by reading, compiled, not run on hardware
//
#include <hip/hip_runtime.h>
#include <hip/hip_bf16.h>
#include <math.h>

typedef int v8i __attribute__((ext_vector_type(8)));

#define BN_EPS 1e-5f

#if defined(__has_builtin)
#if __has_builtin(__builtin_amdgcn_global_load_async_to_lds_b32)
#define HAVE_ASYNC_LDS 1
#endif
#if __has_builtin(__builtin_amdgcn_s_wait_asynccnt)
#define HAVE_WAIT_ASYNC_BUILTIN 1
#endif
#endif

typedef __attribute__((address_space(1))) int as1_int;
typedef __attribute__((address_space(3))) int as3_int;

__device__ __forceinline__ float sigf(float v){ return 1.0f/(1.0f+expf(-v)); }
__device__ __forceinline__ unsigned char umax8(unsigned char a, unsigned char b){ return a>b?a:b; }

__global__ void k_zero(unsigned int* __restrict__ p, int nwords){
  int i = blockIdx.x*blockDim.x + threadIdx.x;
  if (i < nwords) p[i] = 0u;
}

__global__ void k_wmax(const float* __restrict__ w, int n, int* __restrict__ out){
  __shared__ int smax;
  if (threadIdx.x == 0) smax = 0;
  __syncthreads();
  int i = blockIdx.x*blockDim.x + threadIdx.x;
  int local = 0;
  if (i < n) local = __float_as_int(fabsf(tanhf(w[i])));
  atomicMax(&smax, local);
  __syncthreads();
  if (threadIdx.x == 0) atomicMax(out, smax);
}

__global__ void k_wpack(const float* __restrict__ w, const int* __restrict__ maxbits,
                        signed char* __restrict__ wp,
                        int Cout, int Cin, int ks, int cinShift, int coutTiles){
  int n = Cout*Cin*ks*ks;
  int i = blockIdx.x*blockDim.x + threadIdx.x;
  if (i >= n) return;
  float mt = __int_as_float(*maxbits);
  float wn = tanhf(w[i]) * (0.5f/mt) + 0.5f;
  int q = (int)rintf(wn * 15.0f);
  q = q < 0 ? 0 : (q > 15 ? 15 : q);
  int wi = 2*q - 15;
  int kk2 = ks*ks;
  int co = i / (Cin*kk2);
  int r  = i - co*(Cin*kk2);
  int ci = r / kk2;
  int tap = r - ci*kk2;
  int kg = (tap << cinShift) + ci;
  int kc = kg >> 6;
  int j  = kg & 63;
  int b  = j & 3;
  int c1 = (j >> 2) & 1;
  int hf = (j >> 4) & 1;
  int b1 = (j >> 3) & 1;
  int a1 = (j >> 5) & 1;
  int v  = 4*a1 + 2*b1 + c1;
  int lane = (co & 15) + 16*hf;
  int cot  = co >> 4;
  long off = (((long)kc*coutTiles + cot)*32 + lane)*32 + v*4 + b;
  wp[off] = (signed char)wi;
}

__global__ void k_bnprep(const float* __restrict__ g, const float* __restrict__ b,
                         const float* __restrict__ m, const float* __restrict__ v,
                         int c, float* __restrict__ scale, float* __restrict__ bias){
  int i = blockIdx.x*blockDim.x + threadIdx.x;
  if (i < c) {
    float inv = g[i] / sqrtf(v[i] + BN_EPS);
    scale[i] = inv * (1.0f/15.0f);
    bias[i]  = (b[i] - m[i]*inv) * 15.0f;
  }
}

__global__ void k_fill_l8(float* __restrict__ scale, float* __restrict__ bias){
  int i = threadIdx.x;
  scale[i] = 1.0f/225.0f;
  bias[i]  = 0.0f;
}

__global__ void k_qin(const float* __restrict__ x, unsigned char* __restrict__ a0,
                      int N, int H, int W){
  int hw = H*W;
  int tot = N*hw;
  int i = blockIdx.x*blockDim.x + threadIdx.x;
  if (i >= tot) return;
  int n = i / hw;
  int r = i - n*hw;
  unsigned int wvv = 0u;
  #pragma unroll
  for (int c = 0; c < 3; ++c) {
    float f = x[((long)n*3 + c)*hw + r];
    f = fminf(fmaxf(f, 0.0f), 1.0f);
    int q = (int)rintf(f * 15.0f);
    wvv |= ((unsigned int)q) << (8*c);
  }
  ((unsigned int*)a0)[i] = wvv;
}

__global__ void k_pool(const unsigned char* __restrict__ in, unsigned char* __restrict__ out,
                       int N, int H, int W, int C){
  int Ho = H >> 1, Wo = W >> 1;
  long tot = (long)N*Ho*Wo*C;
  long i = (long)blockIdx.x*blockDim.x + threadIdx.x;
  if (i >= tot) return;
  int c = (int)(i % C);
  long t = i / C;
  int x2 = (int)(t % Wo); t /= Wo;
  int y2 = (int)(t % Ho);
  int n  = (int)(t / Ho);
  const unsigned char* p = in + (((long)n*H + 2*y2)*W + 2*x2)*C + c;
  long rs = (long)W*C;
  out[i] = umax8(umax8(p[0], p[C]), umax8(p[rs], p[rs+C]));
}

__global__ __launch_bounds__(256)
void k_qconv(const unsigned char* __restrict__ actIn,
             const signed char*  __restrict__ wpack,
             const float* __restrict__ scale,
             const float* __restrict__ bias,
             unsigned char* __restrict__ actOut,
             float* __restrict__ praw,
             const unsigned char* __restrict__ zeroPage,
             int N, int H, int W, int cinShift, int taps,
             int Kchunks, int coutTiles, int Cout, int quantOut)
{
  __shared__ __align__(32) unsigned int Bs[4][32][8];
  const int tid  = threadIdx.x;
  const int lane = tid & 31;
  const int wv   = tid >> 5;
  const int mt   = wv >> 1;
  const int ng   = wv & 1;
  const int hw   = H*W;
  const int totalPix = N*hw;
  const int pixBase  = blockIdx.x * 64;
  const int Ktot  = taps << cinShift;
  const int CoutP = coutTiles << 4;

  const int sl    = tid >> 3;
  const int sv    = tid & 7;
  const int shalf = sl >> 4;
  const int sncol = sl & 15;
  const int sj = ((sv >> 2) << 5) + (shalf << 4) + ((sv & 3) << 2);

  int  py[4], px[4];
  long prow[4];
  bool pval[4];
  #pragma unroll
  for (int k = 0; k < 4; ++k) {
    int p = pixBase + k*16 + sncol;
    pval[k] = (p < totalPix);
    int pc = pval[k] ? p : 0;
    int nimg = pc / hw;
    int rem  = pc - nimg*hw;
    int y = rem / W;
    py[k] = y;
    px[k] = rem - y*W;
    prow[k] = (long)nimg * hw;
  }

  v8i acc0 = {0,0,0,0,0,0,0,0};
  v8i acc1 = {0,0,0,0,0,0,0,0};

  for (int kc = 0; kc < Kchunks; ++kc) {
    __syncthreads();
    int kg  = (kc << 6) + sj;
    int tap = kg >> cinShift;
    int ci  = kg & ((1 << cinShift) - 1);
    int dy = 0, dx = 0;
    if (taps == 9) { int q3 = tap/3; dy = q3 - 1; dx = tap - 3*q3 - 1; }
    bool kok = (kg < Ktot);
    #pragma unroll
    for (int k = 0; k < 4; ++k) {
      int yy = py[k] + dy;
      int xx = px[k] + dx;
      bool ok = kok & pval[k] & ((unsigned)yy < (unsigned)H) & ((unsigned)xx < (unsigned)W);
      long a = ((prow[k] + (long)yy*W + xx) << cinShift) + ci;
      const unsigned char* g = ok ? (actIn + a) : zeroPage;
#ifdef HAVE_ASYNC_LDS
      __builtin_amdgcn_global_load_async_to_lds_b32(
          (as1_int*)(unsigned char*)g,
          (as3_int*)&Bs[k][sl][sv],
          0, 0);
#else
      Bs[k][sl][sv] = *(const unsigned int*)g;
#endif
    }
#ifdef HAVE_ASYNC_LDS
#ifdef HAVE_WAIT_ASYNC_BUILTIN
    __builtin_amdgcn_s_wait_asynccnt(0);
#else
    asm volatile("s_wait_asynccnt 0" ::: "memory");
#endif
#endif
    __syncthreads();

    if (mt < coutTiles) {
      const signed char* ap = wpack + (((long)kc*coutTiles + mt)*32 + lane)*32;
      if (kc + 1 < Kchunks)
        __builtin_prefetch(ap + (long)coutTiles*1024, 0, 1);
      v8i afrag = *(const v8i*)ap;
      v8i b0 = *(const v8i*)&Bs[ng*2 + 0][lane][0];
      v8i b1 = *(const v8i*)&Bs[ng*2 + 1][lane][0];
      acc0 = __builtin_amdgcn_wmma_i32_16x16x64_iu8(true, afrag, false, b0, acc0, false, false);
      acc1 = __builtin_amdgcn_wmma_i32_16x16x64_iu8(true, afrag, false, b1, acc1, false, false);
    }
  }

  if (mt >= coutTiles) return;
  int ncol  = lane & 15;
  int rhalf = lane >> 4;
  #pragma unroll
  for (int s = 0; s < 2; ++s) {
    v8i acc = s ? acc1 : acc0;
    int p = pixBase + (ng*2 + s)*16 + ncol;
    if (p >= totalPix) continue;
    if (quantOut) {
      unsigned char* dst = actOut + (long)p * CoutP;
      #pragma unroll
      for (int r = 0; r < 8; ++r) {
        int co = (mt << 4) + rhalf*8 + r;
        float yf = (float)acc[r] * scale[co] + bias[co];
        int q = (int)rintf(yf);
        q = q < 0 ? 0 : (q > 15 ? 15 : q);
        dst[co] = (unsigned char)q;
      }
    } else {
      int nimg = p / hw;
      int rem  = p - nimg*hw;
      #pragma unroll
      for (int r = 0; r < 8; ++r) {
        int co = (mt << 4) + rhalf*8 + r;
        if (co < Cout)
          praw[((long)nimg*Cout + co)*hw + rem] = (float)acc[r] * scale[co] + bias[co];
      }
    }
  }
}

__global__ void k_decode(const float* __restrict__ praw, float* __restrict__ out, long half){
  int i = blockIdx.x*blockDim.x + threadIdx.x;
  if (i >= 16*4800) return;
  int x = i % 40;
  int y = (i/40) % 20;
  int a = (i/800) % 6;
  int bimg = i / 4800;
  float v[6];
  #pragma unroll
  for (int o = 0; o < 6; ++o)
    v[o] = praw[(((long)bimg*36 + a*6 + o)*20 + y)*40 + x];
  long base = (long)i*6;
  out[base+0] = (sigf(v[0]) + (float)x) * 16.0f;
  out[base+1] = (sigf(v[1]) + (float)y) * 16.0f;
  out[base+2] = expf(v[2]) * 20.0f;
  out[base+3] = expf(v[3]) * 20.0f;
  out[base+4] = sigf(v[4]);
  out[base+5] = sigf(v[5]);
  #pragma unroll
  for (int o = 0; o < 6; ++o)
    out[half + base + o] = v[o];
}

extern "C" void kernel_launch(void* const* d_in, const int* in_sizes, int n_in,
                              void* d_out, int out_size, void* d_ws, size_t ws_size,
                              hipStream_t stream)
{
  (void)in_sizes; (void)n_in; (void)ws_size;
  const float* x = (const float*)d_in[0];
  const float* wgt[9];
  for (int i = 0; i < 9; ++i) wgt[i] = (const float*)d_in[1+i];

  struct LC { int H,W,cinShift,taps,Kc,cot,Cout,pool,Cin,ks; };
  static const LC L[9] = {
    {320,640,2,9,1,1,16,1, 3,3},
    {160,320,4,9,3,2,32,1,16,3},
    { 80,160,5,9,5,4,64,1,32,3},
    { 40, 80,6,9,9,4,64,1,64,3},
    { 20, 40,6,9,9,4,64,0,64,3},
    { 20, 40,6,9,9,4,64,0,64,3},
    { 20, 40,6,9,9,4,64,0,64,3},
    { 20, 40,6,9,9,4,64,0,64,3},
    { 20, 40,6,1,1,3,36,0,64,1},
  };

  char* ws = (char*)d_ws;
  size_t off = 0;
  auto al = [&](size_t bytes)->size_t{
    size_t o = off; off = (off + bytes + 255) & ~(size_t)255; return o; };

  size_t wpOff[9];
  for (int i = 0; i < 9; ++i) wpOff[i] = al((size_t)L[i].Kc*L[i].cot*1024);
  size_t maxOff = al(9*sizeof(int));
  size_t sbOff  = al(9*128*sizeof(float));
  size_t zpOff  = al(256);
  size_t zeroBytes = off;
  size_t prawOff = al((size_t)16*36*800*sizeof(float));
  size_t bufPOff = al((size_t)16*320*640*16);
  size_t bufAOff = al((size_t)16*320*640*4);
  size_t bufBOff = al((size_t)16*320*640*4);

  int zw = (int)(zeroBytes/4);
  k_zero<<<(zw+255)/256,256,0,stream>>>((unsigned int*)ws, zw);

  for (int i = 0; i < 9; ++i) {
    int n = L[i].Cout*L[i].Cin*L[i].ks*L[i].ks;
    k_wmax<<<(n+255)/256,256,0,stream>>>(wgt[i], n, (int*)(ws+maxOff)+i);
  }
  for (int i = 0; i < 9; ++i) {
    int n = L[i].Cout*L[i].Cin*L[i].ks*L[i].ks;
    k_wpack<<<(n+255)/256,256,0,stream>>>(wgt[i], (const int*)(ws+maxOff)+i,
        (signed char*)(ws+wpOff[i]), L[i].Cout, L[i].Cin, L[i].ks, L[i].cinShift, L[i].cot);
  }

  for (int i = 0; i < 8; ++i) {
    const float* g = (const float*)d_in[10+4*i+0];
    const float* b = (const float*)d_in[10+4*i+1];
    const float* m = (const float*)d_in[10+4*i+2];
    const float* v = (const float*)d_in[10+4*i+3];
    float* sc = (float*)(ws + sbOff) + i*128;
    k_bnprep<<<1,64,0,stream>>>(g, b, m, v, L[i].Cout, sc, sc+64);
  }
  { float* sc = (float*)(ws + sbOff) + 8*128;
    k_fill_l8<<<1,64,0,stream>>>(sc, sc+64); }

  { int tot = 16*320*640;
    k_qin<<<(tot+255)/256,256,0,stream>>>(x, (unsigned char*)(ws+bufAOff), 16, 320, 640); }

  char* bufs[3] = { ws+bufAOff, ws+bufBOff, ws+bufPOff };
  static const int inIdx[9]   = {0,1,0,1,0,1,0,1,0};
  static const int outIdx[9]  = {2,2,2,2,1,0,1,0,-1};
  static const int poolDst[9] = {1,0,1,0,-1,-1,-1,-1,-1};
  const unsigned char* zp = (const unsigned char*)(ws + zpOff);
  for (int i = 0; i < 9; ++i) {
    const LC& c = L[i];
    int P = 16*c.H*c.W;
    int blocks = (P + 63)/64;
    float* sc = (float*)(ws+sbOff) + i*128;
    unsigned char* aout = (i == 8) ? nullptr : (unsigned char*)bufs[outIdx[i]];
    float* pr = (i == 8) ? (float*)(ws+prawOff) : nullptr;
    k_qconv<<<blocks,256,0,stream>>>(
        (const unsigned char*)bufs[inIdx[i]],
        (const signed char*)(ws+wpOff[i]),
        sc, sc+64, aout, pr, zp,
        16, c.H, c.W, c.cinShift, c.taps, c.Kc, c.cot, c.Cout, (i == 8) ? 0 : 1);
    if (c.pool) {
      long tot = (long)16*(c.H/2)*(c.W/2)*c.Cout;
      int pb = (int)((tot + 255)/256);
      k_pool<<<pb,256,0,stream>>>((const unsigned char*)bufs[outIdx[i]],
                                  (unsigned char*)bufs[poolDst[i]],
                                  16, c.H, c.W, c.Cout);
    }
  }

  { long half = (long)out_size/2;
    int tot = 16*4800;
    k_decode<<<(tot+255)/256,256,0,stream>>>((const float*)(ws+prawOff), (float*)d_out, half); }
}
